// ViTForClassification_63634235458031
// MI455X (gfx1250) — matrix operand order repaired and checked by reading, not compiled, not run on hardware
//
#include <hip/hip_runtime.h>
#include <hip/hip_bf16.h>

// ---------------------------------------------------------------------------
// ViT-B/16 forward for MI455X (gfx1250, wave32, WMMA).
// All dense math via v_wmma_f32_16x16x32_f16 (f16 in, f32 accum).
// LDS tiles are stored in WMMA *fragment layout* so every fragment load is a
// single per-lane 32B read (2x ds_load_b128), no repacking VALU.
// Token-activation buffers are padded to M_PAD (multiple of 64) so the main
// GEMM is completely guard-free (no clamps, no selects, no store guards):
// pad-row garbage is row-local through the GEMM chain and never consumed.
// ---------------------------------------------------------------------------

#define VIT_B   32
#define VIT_S   197
#define VIT_D   768
#define VIT_H   12
#define VIT_DH  64
#define VIT_FF  3072
#define VIT_L   12
#define VIT_NC  14
#define VIT_NT  (VIT_B * VIT_S)    // 6304 real tokens
#define VIT_MP  6336               // padded token rows (99 * 64)
#define VIT_NP  (VIT_B * 196)      // 6272 patches (= 98 * 64, already padded)

typedef __attribute__((ext_vector_type(16))) _Float16 v16h;
typedef __attribute__((ext_vector_type(8)))  float    v8f;

// ---------------------------------------------------------------------------
// WMMA fragment layouts (CDNA5 ISA §7.12.2, wave32), 16x16x32 f16:
//  A 16x32 (code's contiguous k order): element (m,k) -> lane = (m&15) | ((k>>4)<<4), idx = k&15
//  B 32x16 (same order, not the ISA table's): element (k,n) -> lane = (n&15) | ((k>>4)<<4), idx = k&15
//  C/D 16x16 f32: vgpr r -> row M = (lane>>4)*8 + r, col N = lane&15
// LDS tile buffers hold 4 fragments: [tile 0..3][lane 0..31][elem 0..15] f16.
// ---------------------------------------------------------------------------

__device__ __forceinline__ v8f wmma_f16(v16h a, v16h b, v8f c) {
    return __builtin_amdgcn_wmma_f32_16x16x32_f16(false, a, false, b, (short)0, c, false, false);
}

__device__ __forceinline__ float gelu_f(float v) {
    return 0.5f * v * (1.0f + erff(v * 0.70710678118654752f));
}

// Fetch a 64x32 f32 tile. RG: guard/clamp rows vs Rows. KG: guard cols vs
// Keff (scalar loads; slight in-buffer over-read is safe for our P buffer).
template <bool RG, bool KG>
__device__ __forceinline__ void fetch_rows(float4* rr, const float* __restrict__ X,
                                           int ldx, int Rows, int Keff,
                                           int r0, int k0, int tid)
{
#pragma unroll
    for (int it = 0; it < 4; ++it) {
        int idx = tid + it * 128;
        int r = idx >> 3, c4 = (idx & 7) * 4;
        int gr = r0 + r;
        int grc = (RG && gr >= Rows) ? Rows - 1 : gr;
        const float* p = X + (size_t)grc * ldx + k0 + c4;
        float4 v;
        if (KG) {
            int kk = k0 + c4;
            float x0 = p[0], x1 = p[1], x2 = p[2], x3 = p[3];
            v.x = (kk + 0 < Keff) ? x0 : 0.f;
            v.y = (kk + 1 < Keff) ? x1 : 0.f;
            v.z = (kk + 2 < Keff) ? x2 : 0.f;
            v.w = (kk + 3 < Keff) ? x3 : 0.f;
        } else {
            v = *(const float4*)p;
        }
        if (RG && gr >= Rows) v = make_float4(0.f, 0.f, 0.f, 0.f);
        rr[it] = v;
    }
}

// Store fetched 64x32 tile as A-fragments (4 m-tiles of 16 rows).
__device__ __forceinline__ void store_a(_Float16* __restrict__ Asf,
                                        const float4* rr, int tid)
{
#pragma unroll
    for (int it = 0; it < 4; ++it) {
        int idx = tid + it * 128;
        int r = idx >> 3, c4 = (idx & 7) * 4;
        int mt    = r >> 4;
        int lslot = (r & 15) | (((c4 >> 4) & 1) << 4);
        int e0    = c4 & 15;
        _Float16* dst = Asf + ((mt * 32 + lslot) << 4) + e0;
        dst[0] = (_Float16)rr[it].x; dst[1] = (_Float16)rr[it].y;
        dst[2] = (_Float16)rr[it].z; dst[3] = (_Float16)rr[it].w;
    }
}

// Store fetched 64x32 tile (rows = N dim of W[N,K]) as B-fragments
// (4 n-tiles of 16 cols): B element (k,n) with n = row, k = col.
__device__ __forceinline__ void store_b_t(_Float16* __restrict__ Bsf,
                                          const float4* rr, int tid)
{
#pragma unroll
    for (int it = 0; it < 4; ++it) {
        int idx = tid + it * 128;
        int n = idx >> 3, c4 = (idx & 7) * 4;
        int t     = n >> 4;
        int lslot = (n & 15) | (((c4 >> 4) & 1) << 4);
        int e0    = c4 & 15;
        _Float16* dst = Bsf + ((t * 32 + lslot) << 4) + e0;
        dst[0] = (_Float16)rr[it].x; dst[1] = (_Float16)rr[it].y;
        dst[2] = (_Float16)rr[it].z; dst[3] = (_Float16)rr[it].w;
    }
}

// Fetch 32x64 V tile (rows = K dim, guarded vs Keff; cols n 0..63) for P.V.
__device__ __forceinline__ void fetch_v(float4* rr, const float* __restrict__ V,
                                        int ldv, int Keff, int k0, int tid)
{
#pragma unroll
    for (int it = 0; it < 4; ++it) {
        int idx = tid + it * 128;
        int kr = idx >> 4, c4 = (idx & 15) * 4;
        int gk = k0 + kr;
        int gkc = gk < Keff ? gk : Keff - 1;
        float4 v = *(const float4*)(V + (size_t)gkc * ldv + c4);
        if (gk >= Keff) v = make_float4(0.f, 0.f, 0.f, 0.f);
        rr[it] = v;
    }
}

// Store V tile as B-fragments: element (k=kr, n=c4+j).
__device__ __forceinline__ void store_b_nt(_Float16* __restrict__ Bsf,
                                           const float4* rr, int tid)
{
#pragma unroll
    for (int it = 0; it < 4; ++it) {
        int idx = tid + it * 128;
        int kr = idx >> 4, c4 = (idx & 15) * 4;
        int khw = (kr >> 4) << 4;
        int e   = kr & 15;
        int t   = c4 >> 4;
        float vv[4] = { rr[it].x, rr[it].y, rr[it].z, rr[it].w };
#pragma unroll
        for (int j = 0; j < 4; ++j) {
            int n = c4 + j;
            Bsf[((t * 32 + ((n & 15) | khw)) << 4) + e] = (_Float16)vv[j];
        }
    }
}

// ---------------------------------------------------------------------------
// Guard-free WMMA GEMM: out[M,N] = X[M,K] * W[N,K]^T + bias (+res / GELU).
// Requires M % 64 == 0 (padded buffers), N % 64 == 0, K % 32 == 0.
// Block 128 = 4 waves; tile 64x64; wave owns 32x32; K step 32, double-buffered.
// EPI: 0 = bias, 1 = bias + residual, 2 = gelu(bias + acc)
// ---------------------------------------------------------------------------
template <int EPI>
__global__ __launch_bounds__(128) void gemm_kernel(
    const float* __restrict__ X, const float* __restrict__ W,
    const float* __restrict__ bias, const float* __restrict__ res,
    float* __restrict__ out, int M, int N, int K)
{
    __shared__ __align__(32) _Float16 Asf[4 * 32 * 16];
    __shared__ __align__(32) _Float16 Bsf[4 * 32 * 16];

    const int tid  = threadIdx.x;
    const int wave = tid >> 5, lane = tid & 31;
    const int m0 = blockIdx.y * 64, n0 = blockIdx.x * 64;
    const int mw = (wave >> 1) * 32, nw = (wave & 1) * 32;
    const v16h* Af = (const v16h*)Asf;
    const v16h* Bf = (const v16h*)Bsf;

    v8f acc[2][2] = {};
    float4 ra[4], rb[4];
    const int nK = K >> 5;

    fetch_rows<false, false>(ra, X, K, M, K, m0, 0, tid);
    fetch_rows<false, false>(rb, W, K, N, K, n0, 0, tid);

    for (int kt = 0; kt < nK; ++kt) {
        store_a(Asf, ra, tid);
        store_b_t(Bsf, rb, tid);
        __syncthreads();
        if (kt + 1 < nK) {
            fetch_rows<false, false>(ra, X, K, M, K, m0, (kt + 1) << 5, tid);
            fetch_rows<false, false>(rb, W, K, N, K, n0, (kt + 1) << 5, tid);
        }
        v16h a0 = Af[((mw >> 4) + 0) * 32 + lane];
        v16h a1 = Af[((mw >> 4) + 1) * 32 + lane];
        v16h b0 = Bf[((nw >> 4) + 0) * 32 + lane];
        v16h b1 = Bf[((nw >> 4) + 1) * 32 + lane];
        acc[0][0] = wmma_f16(a0, b0, acc[0][0]);
        acc[0][1] = wmma_f16(a0, b1, acc[0][1]);
        acc[1][0] = wmma_f16(a1, b0, acc[1][0]);
        acc[1][1] = wmma_f16(a1, b1, acc[1][1]);
        __syncthreads();
    }

    const int mbase = m0 + mw + ((lane >> 4) * 8);
    const int nbase = n0 + nw + (lane & 15);
#pragma unroll
    for (int nt = 0; nt < 2; ++nt) {
        int col = nbase + nt * 16;
        float bc = bias[col];
#pragma unroll
        for (int mt = 0; mt < 2; ++mt)
#pragma unroll
            for (int r = 0; r < 8; ++r) {
                int row = mbase + mt * 16 + r;
                float v = acc[mt][nt][r] + bc;
                if (EPI == 1) v += res[(size_t)row * N + col];
                if (EPI == 2) v = gelu_f(v);
                out[(size_t)row * N + col] = v;
            }
    }
}

// ---------------------------------------------------------------------------
// Attention scores: scores[bh][S][S] = Q . K^T / 8 (K dim = 64, 2 steps).
// grid (4, 4, B*H), block 128.
// ---------------------------------------------------------------------------
__global__ __launch_bounds__(128) void attn_scores_kernel(
    const float* __restrict__ q, const float* __restrict__ k,
    float* __restrict__ scores)
{
    __shared__ __align__(32) _Float16 Asf[4 * 32 * 16];
    __shared__ __align__(32) _Float16 Bsf[4 * 32 * 16];

    const int tid  = threadIdx.x;
    const int wave = tid >> 5, lane = tid & 31;
    const int bh = blockIdx.z;
    const int b = bh / VIT_H, hh = bh % VIT_H;
    const float* Q  = q + (size_t)b * VIT_S * VIT_D + hh * VIT_DH;
    const float* Km = k + (size_t)b * VIT_S * VIT_D + hh * VIT_DH;
    float* out = scores + (size_t)bh * VIT_S * VIT_S;

    const int m0 = blockIdx.y * 64, n0 = blockIdx.x * 64;
    const int mw = (wave >> 1) * 32, nw = (wave & 1) * 32;
    const v16h* Af = (const v16h*)Asf;
    const v16h* Bf = (const v16h*)Bsf;

    v8f acc[2][2] = {};
    float4 ra[4], rb[4];

    fetch_rows<true, false>(ra, Q,  VIT_D, VIT_S, VIT_DH, m0, 0, tid);
    fetch_rows<true, false>(rb, Km, VIT_D, VIT_S, VIT_DH, n0, 0, tid);

#pragma unroll
    for (int kt = 0; kt < 2; ++kt) {
        store_a(Asf, ra, tid);
        store_b_t(Bsf, rb, tid);
        __syncthreads();
        if (kt == 0) {
            fetch_rows<true, false>(ra, Q,  VIT_D, VIT_S, VIT_DH, m0, 32, tid);
            fetch_rows<true, false>(rb, Km, VIT_D, VIT_S, VIT_DH, n0, 32, tid);
        }
        v16h a0 = Af[((mw >> 4) + 0) * 32 + lane];
        v16h a1 = Af[((mw >> 4) + 1) * 32 + lane];
        v16h b0 = Bf[((nw >> 4) + 0) * 32 + lane];
        v16h b1 = Bf[((nw >> 4) + 1) * 32 + lane];
        acc[0][0] = wmma_f16(a0, b0, acc[0][0]);
        acc[0][1] = wmma_f16(a0, b1, acc[0][1]);
        acc[1][0] = wmma_f16(a1, b0, acc[1][0]);
        acc[1][1] = wmma_f16(a1, b1, acc[1][1]);
        __syncthreads();
    }

    const int mbase = m0 + mw + ((lane >> 4) * 8);
    const int nbase = n0 + nw + (lane & 15);
#pragma unroll
    for (int mt = 0; mt < 2; ++mt)
#pragma unroll
        for (int nt = 0; nt < 2; ++nt)
#pragma unroll
            for (int r = 0; r < 8; ++r) {
                int row = mbase + mt * 16 + r;
                int col = nbase + nt * 16;
                if (row < VIT_S && col < VIT_S)
                    out[(size_t)row * VIT_S + col] = acc[mt][nt][r] * 0.125f;
            }
}

// ---------------------------------------------------------------------------
// O = P . V per (b,h): P [S,S] stride S, V [S,64] stride D, K = 197 padded.
// grid (4 m-tiles, B*H), block 128. N = 64 (single block tile).
// ---------------------------------------------------------------------------
__global__ __launch_bounds__(128) void attn_av_kernel(
    const float* __restrict__ probs, const float* __restrict__ v,
    float* __restrict__ o)
{
    __shared__ __align__(32) _Float16 Asf[4 * 32 * 16];
    __shared__ __align__(32) _Float16 Bsf[4 * 32 * 16];

    const int tid  = threadIdx.x;
    const int wave = tid >> 5, lane = tid & 31;
    const int bh = blockIdx.y;
    const int b = bh / VIT_H, hh = bh % VIT_H;
    const float* P = probs + (size_t)bh * VIT_S * VIT_S;
    const float* V = v + (size_t)b * VIT_S * VIT_D + hh * VIT_DH;
    float* O = o + (size_t)b * VIT_S * VIT_D + hh * VIT_DH;

    const int m0 = blockIdx.x * 64;
    const int mw = (wave >> 1) * 32, nw = (wave & 1) * 32;
    const v16h* Af = (const v16h*)Asf;
    const v16h* Bf = (const v16h*)Bsf;

    v8f acc[2][2] = {};
    float4 ra[4], rb[4];
    const int nK = (VIT_S + 31) >> 5;   // 7

    fetch_rows<true, true>(ra, P, VIT_S, VIT_S, VIT_S, m0, 0, tid);
    fetch_v(rb, V, VIT_D, VIT_S, 0, tid);

    for (int kt = 0; kt < nK; ++kt) {
        store_a(Asf, ra, tid);
        store_b_nt(Bsf, rb, tid);
        __syncthreads();
        if (kt + 1 < nK) {
            fetch_rows<true, true>(ra, P, VIT_S, VIT_S, VIT_S, m0, (kt + 1) << 5, tid);
            fetch_v(rb, V, VIT_D, VIT_S, (kt + 1) << 5, tid);
        }
        v16h a0 = Af[((mw >> 4) + 0) * 32 + lane];
        v16h a1 = Af[((mw >> 4) + 1) * 32 + lane];
        v16h b0 = Bf[((nw >> 4) + 0) * 32 + lane];
        v16h b1 = Bf[((nw >> 4) + 1) * 32 + lane];
        acc[0][0] = wmma_f16(a0, b0, acc[0][0]);
        acc[0][1] = wmma_f16(a0, b1, acc[0][1]);
        acc[1][0] = wmma_f16(a1, b0, acc[1][0]);
        acc[1][1] = wmma_f16(a1, b1, acc[1][1]);
        __syncthreads();
    }

    const int mbase = m0 + mw + ((lane >> 4) * 8);
    const int nbase = nw + (lane & 15);
#pragma unroll
    for (int mt = 0; mt < 2; ++mt)
#pragma unroll
        for (int nt = 0; nt < 2; ++nt)
#pragma unroll
            for (int r = 0; r < 8; ++r) {
                int row = mbase + mt * 16 + r;
                int col = nbase + nt * 16;     // < 64 always
                if (row < VIT_S)
                    O[(size_t)row * VIT_D + col] = acc[mt][nt][r];
            }
}

// ---------------------------------------------------------------------------
// Row softmax over scores[B*H*S][S]; one wave32 per row, in place.
// ---------------------------------------------------------------------------
__global__ __launch_bounds__(32) void softmax_kernel(float* __restrict__ p0)
{
    float* p = p0 + (size_t)blockIdx.x * VIT_S;
    int lane = threadIdx.x;
    float mx = -3.4e38f;
    for (int i = lane; i < VIT_S; i += 32) mx = fmaxf(mx, p[i]);
#pragma unroll
    for (int o = 16; o; o >>= 1) mx = fmaxf(mx, __shfl_xor(mx, o, 32));
    float sum = 0.f;
    for (int i = lane; i < VIT_S; i += 32) { float e = __expf(p[i] - mx); p[i] = e; sum += e; }
#pragma unroll
    for (int o = 16; o; o >>= 1) sum += __shfl_xor(sum, o, 32);
    float inv = 1.f / sum;
    for (int i = lane; i < VIT_S; i += 32) p[i] *= inv;
}

// ---------------------------------------------------------------------------
// LayerNorm over D=768 per token; one 256-thread block per token.
// ---------------------------------------------------------------------------
__global__ __launch_bounds__(256) void layernorm_kernel(
    const float* __restrict__ in, const float* __restrict__ w,
    const float* __restrict__ b, float* __restrict__ out)
{
    __shared__ float red[8];
    const float* x = in + (size_t)blockIdx.x * VIT_D;
    int tid = threadIdx.x, lane = tid & 31, wv = tid >> 5;
    float v0 = x[tid], v1 = x[tid + 256], v2 = x[tid + 512];

    float s = v0 + v1 + v2;
#pragma unroll
    for (int o = 16; o; o >>= 1) s += __shfl_xor(s, o, 32);
    if (lane == 0) red[wv] = s;
    __syncthreads();
    float tot = 0.f;
#pragma unroll
    for (int i = 0; i < 8; ++i) tot += red[i];
    float mean = tot * (1.f / (float)VIT_D);
    __syncthreads();

    float d0 = v0 - mean, d1 = v1 - mean, d2 = v2 - mean;
    s = d0 * d0 + d1 * d1 + d2 * d2;
#pragma unroll
    for (int o = 16; o; o >>= 1) s += __shfl_xor(s, o, 32);
    if (lane == 0) red[wv] = s;
    __syncthreads();
    tot = 0.f;
#pragma unroll
    for (int i = 0; i < 8; ++i) tot += red[i];
    float rstd = rsqrtf(tot * (1.f / (float)VIT_D) + 1e-5f);

    out[(size_t)blockIdx.x * VIT_D + tid]       = d0 * rstd * w[tid]       + b[tid];
    out[(size_t)blockIdx.x * VIT_D + tid + 256] = d1 * rstd * w[tid + 256] + b[tid + 256];
    out[(size_t)blockIdx.x * VIT_D + tid + 512] = d2 * rstd * w[tid + 512] + b[tid + 512];
}

// ---------------------------------------------------------------------------
// im2col: Xp[b*196+p][c*256+py*16+px] = x[b][c][gy*16+py][gx*16+px]
// ---------------------------------------------------------------------------
__global__ __launch_bounds__(256) void im2col_kernel(
    const float* __restrict__ x, float* __restrict__ xp)
{
    int r = blockIdx.x;                   // 0..6271
    int b = r / 196, p = r % 196;
    int gy = p / 14, gx = p % 14;
    float* dst = xp + (size_t)r * VIT_D;
    for (int i = threadIdx.x; i < VIT_D; i += 256) {
        int c = i >> 8, py = (i >> 4) & 15, px = i & 15;
        dst[i] = x[(((size_t)b * 3 + c) * 224 + gy * 16 + py) * 224 + gx * 16 + px];
    }
}

// ---------------------------------------------------------------------------
// h[b][0] = cls + pos[0]; h[b][s] = emb[b*196+s-1] + pos[s]
// ---------------------------------------------------------------------------
__global__ __launch_bounds__(256) void assemble_kernel(
    const float* __restrict__ emb, const float* __restrict__ cls,
    const float* __restrict__ pos, float* __restrict__ h)
{
    int s = blockIdx.x, b = blockIdx.y;
    float* hr = h + ((size_t)b * VIT_S + s) * VIT_D;
    const float* pr = pos + (size_t)s * VIT_D;
    const float* src = (s == 0) ? cls : emb + ((size_t)b * 196 + (s - 1)) * VIT_D;
    for (int i = threadIdx.x; i < VIT_D; i += 256)
        hr[i] = src[i] + pr[i];
}

// ---------------------------------------------------------------------------
// logits[b][c] = h[b][0] . clf_w[c] + clf_b[c]
// ---------------------------------------------------------------------------
__global__ __launch_bounds__(64) void classifier_kernel(
    const float* __restrict__ h, const float* __restrict__ w,
    const float* __restrict__ bias, float* __restrict__ out)
{
    int idx = blockIdx.x * 64 + threadIdx.x;
    if (idx >= VIT_B * VIT_NC) return;
    int b = idx / VIT_NC, c = idx % VIT_NC;
    const float* hr = h + (size_t)b * VIT_S * VIT_D;  // token 0
    const float* wr = w + (size_t)c * VIT_D;
    float s = bias[c];
    for (int i = 0; i < VIT_D; ++i) s += hr[i] * wr[i];
    out[idx] = s;
}

// ---------------------------------------------------------------------------
extern "C" void kernel_launch(void* const* d_in, const int* in_sizes, int n_in,
                              void* d_out, int out_size, void* d_ws, size_t ws_size,
                              hipStream_t stream)
{
    const float* x       = (const float*)d_in[0];
    const float* patch_w = (const float*)d_in[1];
    const float* patch_b = (const float*)d_in[2];
    const float* cls     = (const float*)d_in[3];
    const float* pos     = (const float*)d_in[4];
    const float* ln1_w   = (const float*)d_in[5];
    const float* ln1_b   = (const float*)d_in[6];
    const float* wq = (const float*)d_in[7];  const float* bq = (const float*)d_in[8];
    const float* wk = (const float*)d_in[9];  const float* bk = (const float*)d_in[10];
    const float* wv = (const float*)d_in[11]; const float* bv = (const float*)d_in[12];
    const float* wo = (const float*)d_in[13]; const float* bo = (const float*)d_in[14];
    const float* ln2_w = (const float*)d_in[15]; const float* ln2_b = (const float*)d_in[16];
    const float* fc1_w = (const float*)d_in[17]; const float* fc1_b = (const float*)d_in[18];
    const float* fc2_w = (const float*)d_in[19]; const float* fc2_b = (const float*)d_in[20];
    const float* clf_w = (const float*)d_in[21]; const float* clf_b = (const float*)d_in[22];
    float* logits = (float*)d_out;

    // Workspace layout (floats), all token matrices padded to VIT_MP rows:
    // h | ln | q | k | v | big
    const size_t TOKP = (size_t)VIT_MP * VIT_D;            // 6336*768
    float* hbuf = (float*)d_ws;
    float* lnb  = hbuf + TOKP;
    float* qb   = lnb  + TOKP;
    float* kb   = qb   + TOKP;
    float* vb   = kb   + TOKP;
    float* big  = vb   + TOKP;          // VIT_MP*3072: xp / scores / ffn1

    const dim3 blk128(128), blk256(256);
    const dim3 gemmD (VIT_D  / 64, VIT_MP / 64);           // (12, 99)
    const dim3 gemmFF(VIT_FF / 64, VIT_MP / 64);           // (48, 99)

    // ---- Patch embedding ----
    im2col_kernel<<<VIT_NP, blk256, 0, stream>>>(x, big);
    gemm_kernel<0><<<dim3(VIT_D / 64, VIT_NP / 64), blk128, 0, stream>>>(
        big, patch_w, patch_b, nullptr, qb, VIT_NP, VIT_D, VIT_D);
    assemble_kernel<<<dim3(VIT_S, VIT_B), blk256, 0, stream>>>(qb, cls, pos, hbuf);

    // ---- Transformer blocks ----
    for (int l = 0; l < VIT_L; ++l) {
        const size_t oD  = (size_t)l * VIT_D;
        const size_t oDD = (size_t)l * VIT_D * VIT_D;
        const size_t oFD = (size_t)l * VIT_FF * VIT_D;
        const size_t oF  = (size_t)l * VIT_FF;

        layernorm_kernel<<<VIT_NT, blk256, 0, stream>>>(hbuf, ln1_w + oD, ln1_b + oD, lnb);
        gemm_kernel<0><<<gemmD, blk128, 0, stream>>>(lnb, wq + oDD, bq + oD, nullptr, qb, VIT_MP, VIT_D, VIT_D);
        gemm_kernel<0><<<gemmD, blk128, 0, stream>>>(lnb, wk + oDD, bk + oD, nullptr, kb, VIT_MP, VIT_D, VIT_D);
        gemm_kernel<0><<<gemmD, blk128, 0, stream>>>(lnb, wv + oDD, bv + oD, nullptr, vb, VIT_MP, VIT_D, VIT_D);

        attn_scores_kernel<<<dim3(4, 4, VIT_B * VIT_H), blk128, 0, stream>>>(qb, kb, big);
        softmax_kernel<<<VIT_B * VIT_H * VIT_S, 32, 0, stream>>>(big);
        attn_av_kernel<<<dim3(4, VIT_B * VIT_H), blk128, 0, stream>>>(big, vb, lnb); // o -> lnb

        gemm_kernel<1><<<gemmD, blk128, 0, stream>>>(lnb, wo + oDD, bo + oD, hbuf, hbuf, VIT_MP, VIT_D, VIT_D);

        layernorm_kernel<<<VIT_NT, blk256, 0, stream>>>(hbuf, ln2_w + oD, ln2_b + oD, qb);
        gemm_kernel<2><<<gemmFF, blk128, 0, stream>>>(qb, fc1_w + oFD, fc1_b + oF, nullptr, big, VIT_MP, VIT_FF, VIT_D);
        gemm_kernel<1><<<gemmD, blk128, 0, stream>>>(big, fc2_w + oFD, fc2_b + oD, hbuf, hbuf, VIT_MP, VIT_D, VIT_FF);
    }

    // ---- Classifier head ----
    classifier_kernel<<<(VIT_B * VIT_NC + 63) / 64, 64, 0, stream>>>(hbuf, clf_w, clf_b, logits);
}
